// SparseDepthTransformer_33930241638432
// MI455X (gfx1250) — compile-verified, matrix operand order checked by reading, not run on hardware
//
#include <hip/hip_runtime.h>
#include <hip/hip_bf16.h>

// SparseDepthTransformer forward for MI455X (gfx1250, wave32).
//
// Roofline: ~30 GFLOP total; dominant byte cost is the 82MB fp32 logits
// write (~3.5us @ 23.3 TB/s). All matmuls (qkv, attention S/O, out-proj,
// FF1/FF2, vocab head) run on v_wmma_f32_16x16x32_f16 with fp32 accumulate.
// Fragment layouts follow the CDNA5 ISA tables so A/B loads are two
// global_load_b128 per lane. Attention is a wave-level flash kernel; softmax
// row reductions use DPP16 ROW_XMASK (pure VALU, no ds_bpermute / dscnt
// stalls), and the P-transpose goes through LDS guarded by s_wait_dscnt.

typedef __attribute__((ext_vector_type(8)))  _Float16 v8h;
typedef __attribute__((ext_vector_type(16))) _Float16 v16h;
typedef __attribute__((ext_vector_type(8)))  float    v8f;

#define NTOK  4096   // B*T
#define DMODEL 64
#define NLAYER 6
#define NVOCAB 5000
#define VPAD   5008  // vocab padded to a multiple of 16 for tile loads

__device__ __forceinline__ v8f wmma_f16(v16h a, v16h b, v8f c) {
  // v_wmma_f32_16x16x32_f16 : D = A(16x32) * B(32x16) + C(16x16 f32)
  return __builtin_amdgcn_wmma_f32_16x16x32_f16(
      /*neg_a=*/false, a, /*neg_b=*/false, b,
      /*c_mod=*/(short)0, c, /*reuse_a=*/false, /*reuse_b=*/false);
}

// DPP16 ROW_XMASK lane swap: lane <-> lane ^ MASK within each 16-lane row.
// VALU-only cross-lane op: no LDS traffic, no DScnt waits.
template <int CTRL>
__device__ __forceinline__ float dppf(float x) {
  union { float f; int i; } u, r;
  u.f = x;
  r.i = __builtin_amdgcn_update_dpp(u.i, u.i, CTRL, 0xf, 0xf, false);
  return r.f;
}
__device__ __forceinline__ float rowmax16(float v) {
  v = fmaxf(v, dppf<0x161>(v));  // ROW_XMASK ^1
  v = fmaxf(v, dppf<0x162>(v));  // ROW_XMASK ^2
  v = fmaxf(v, dppf<0x164>(v));  // ROW_XMASK ^4
  v = fmaxf(v, dppf<0x168>(v));  // ROW_XMASK ^8
  return v;
}
__device__ __forceinline__ float rowsum16(float v) {
  v += dppf<0x161>(v);
  v += dppf<0x162>(v);
  v += dppf<0x164>(v);
  v += dppf<0x168>(v);
  return v;
}

// Load one 16x32 16-bit fragment (A layout; B 32x16 is symmetric with
// "row" = output column). ISA layout: lane l (g=l/16, r=l%16) holds row r,
// K = {8g..8g+7} and {16+8g..16+8g+7}  -> two contiguous b128 loads.
__device__ __forceinline__ v16h ldfrag(const _Float16* __restrict__ base,
                                       int stride, int row0, int k0) {
  const int l = threadIdx.x & 31;
  const int g = l >> 4;
  const int r = l & 15;
  const _Float16* p = base + (size_t)(row0 + r) * stride + k0 + g * 8;
  union { v16h v; v8h h2[2]; } u;
  u.h2[0] = *(const v8h*)p;
  u.h2[1] = *(const v8h*)(p + 16);
  return u.v;
}

// ---------------- elementwise / prep kernels ----------------

__global__ void k_cvt(const float* __restrict__ s, _Float16* __restrict__ d,
                      int n, int nsrc) {
  int i = blockIdx.x * 256 + threadIdx.x;
  if (i < n) d[i] = (_Float16)(i < nsrc ? s[i] : 0.f);
}

__global__ void k_embed(const int* __restrict__ tok,
                        const float* __restrict__ ew,
                        const float* __restrict__ pos,
                        const float* __restrict__ sw,
                        const float* __restrict__ sb,
                        float* __restrict__ x, float* __restrict__ scv) {
  const int n = blockIdx.x, d = threadIdx.x;
  const float v = ew[(size_t)tok[n] * DMODEL + d] +
                  pos[(size_t)(n & 2047) * DMODEL + d];
  x[(size_t)n * DMODEL + d] = v;
  __shared__ float red[DMODEL];
  red[d] = v * sw[d];
  __syncthreads();
  for (int s = 32; s > 0; s >>= 1) {
    if (d < s) red[d] += red[d + s];
    __syncthreads();
  }
  if (d == 0) scv[n] = 1.f / (1.f + __expf(-(red[0] + sb[0])));
}

__global__ void k_ln(const float* __restrict__ in, const float* __restrict__ g,
                     const float* __restrict__ b, _Float16* __restrict__ out) {
  const int n = blockIdx.x, d = threadIdx.x;
  __shared__ float red[DMODEL];
  const float v = in[(size_t)n * DMODEL + d];
  red[d] = v;
  __syncthreads();
  for (int s = 32; s > 0; s >>= 1) {
    if (d < s) red[d] += red[d + s];
    __syncthreads();
  }
  const float mu = red[0] * (1.f / 64.f);
  __syncthreads();
  const float dv = v - mu;
  red[d] = dv * dv;
  __syncthreads();
  for (int s = 32; s > 0; s >>= 1) {
    if (d < s) red[d] += red[d + s];
    __syncthreads();
  }
  const float var = red[0] * (1.f / 64.f);
  out[(size_t)n * DMODEL + d] =
      (_Float16)(dv * rsqrtf(var + 1e-5f) * g[d] + b[d]);
}

// Vt[c][n] = V[n][c] so V fragments for the O=P*V WMMA are contiguous b128s.
__global__ void k_transv(const _Float16* __restrict__ qkv,
                         _Float16* __restrict__ vt) {
  const int n = blockIdx.x, d = threadIdx.x;
  vt[(size_t)d * NTOK + n] = qkv[(size_t)n * 192 + 128 + d];
}

// ---------------- generic WMMA GEMM: C = A[M,K] * W[N,K]^T + bias ----------
// EPI 0: out f16 (qkv)        EPI 1: t = x + out (out-proj residual)
// EPI 2: out f16 relu (ff1)   EPI 3: x = keep ? t + out : x (ff2 + where)
// EPI 4: logits fp32, col-bounded at Nout

template <int EPI>
__global__ void k_gemm(const _Float16* __restrict__ A,
                       const _Float16* __restrict__ Bw,
                       const float* __restrict__ bias,
                       float* __restrict__ outf, _Float16* __restrict__ outh,
                       const float* __restrict__ res,
                       const float* __restrict__ scv, float thresh,
                       int K, int Nout) {
  const int wid = threadIdx.x >> 5;
  const int l = threadIdx.x & 31;
  const int m0 = (blockIdx.x * 4 + wid) * 16;
  const int n0 = blockIdx.y * 16;
  v8f c = {0.f, 0.f, 0.f, 0.f, 0.f, 0.f, 0.f, 0.f};
  for (int kb = 0; kb < K; kb += 32) {
    if (kb + 32 < K)  // hint next weight chunk into cache (global_prefetch_b8)
      __builtin_prefetch(Bw + (size_t)(n0 + (l & 15)) * K + kb + 32, 0, 1);
    v16h a = ldfrag(A, K, m0, kb);
    v16h b = ldfrag(Bw, K, n0, kb);
    c = wmma_f16(a, b, c);
  }
  // C layout: lane holds column n0+(l&15); rows m0 + j + 8*(l>>4).
  const int col = n0 + (l & 15);
  const int rbase = m0 + ((l >> 4) << 3);
  const float bv = (col < Nout) ? bias[col] : 0.f;
#pragma unroll
  for (int j = 0; j < 8; ++j) {
    const int row = rbase + j;
    const float v = c[j] + bv;
    if (EPI == 0) {
      outh[(size_t)row * Nout + col] = (_Float16)v;
    } else if (EPI == 1) {
      outf[(size_t)row * Nout + col] = res[(size_t)row * Nout + col] + v;
    } else if (EPI == 2) {
      outh[(size_t)row * Nout + col] = (_Float16)fmaxf(v, 0.f);
    } else if (EPI == 3) {
      if (scv[row] > thresh)
        outf[(size_t)row * Nout + col] = res[(size_t)row * Nout + col] + v;
    } else {
      if (col < Nout) outf[(size_t)row * Nout + col] = v;
    }
  }
}

// ---------------- flash attention: one wave = 16 query rows x 1 head -------

__global__ void k_attn(const _Float16* __restrict__ qkv,
                       const _Float16* __restrict__ vt,
                       const float* __restrict__ scv, float thresh,
                       _Float16* __restrict__ o) {
  __shared__ __align__(16) _Float16 pbuf[4][16 * 32];
  const int wid = threadIdx.x >> 5;
  const int l = threadIdx.x & 31;
  const int gi = blockIdx.x * 4 + wid;
  const int h = gi & 1;
  const int m0 = (gi >> 1) << 4;
  _Float16* pb = pbuf[wid];

  // Q fragment (A layout) is loop-invariant; fold 1/sqrt(32) into Q so the
  // WMMA result is already scaled (saves 16 v_mul per key chunk).
  v16h aq = ldfrag(qkv + h * 32, 192, m0, 0);
#pragma unroll
  for (int i = 0; i < 16; ++i)
    aq[i] = (_Float16)((float)aq[i] * 0.17677669529663687f);

  float mrun[8], lrun[8];
#pragma unroll
  for (int j = 0; j < 8; ++j) { mrun[j] = -3.0e38f; lrun[j] = 0.f; }
  v8f oc0 = {0.f, 0.f, 0.f, 0.f, 0.f, 0.f, 0.f, 0.f};
  v8f oc1 = {0.f, 0.f, 0.f, 0.f, 0.f, 0.f, 0.f, 0.f};

  for (int kb = 0; kb < NTOK; kb += 32) {
    // S tiles: B = K-matrix rows (keys) as output columns, K-dim = head_dim
    v16h bk0 = ldfrag(qkv + 64 + h * 32, 192, kb, 0);
    v16h bk1 = ldfrag(qkv + 64 + h * 32, 192, kb + 16, 0);
    v8f s0 = {0.f, 0.f, 0.f, 0.f, 0.f, 0.f, 0.f, 0.f};
    v8f s1 = {0.f, 0.f, 0.f, 0.f, 0.f, 0.f, 0.f, 0.f};
    s0 = wmma_f16(aq, bk0, s0);
    s1 = wmma_f16(aq, bk1, s1);

    const bool kp0 = scv[kb + (l & 15)] > thresh;
    const bool kp1 = scv[kb + 16 + (l & 15)] > thresh;
#pragma unroll
    for (int j = 0; j < 8; ++j) {
      s0[j] = kp0 ? s0[j] : -1e9f;
      s1[j] = kp1 ? s1[j] : -1e9f;
    }
    // Online softmax. C-layout row j+8*(l>>4) spans one 16-lane DPP row, so
    // ROW_XMASK reductions keep stats lane-uniform without LDS traffic.
#pragma unroll
    for (int j = 0; j < 8; ++j) {
      const float rm = rowmax16(fmaxf(s0[j], s1[j]));
      const float mnew = fmaxf(mrun[j], rm);
      const float alpha = __expf(mrun[j] - mnew);
      const float p0 = __expf(s0[j] - mnew);
      const float p1 = __expf(s1[j] - mnew);
      const float ps = rowsum16(p0 + p1);
      lrun[j] = lrun[j] * alpha + ps;
      mrun[j] = mnew;
      oc0[j] *= alpha;
      oc1[j] *= alpha;
      const int prow = j + ((l >> 4) << 3);
      pb[prow * 32 + (l & 15)] = (_Float16)p0;        // P in row-major LDS
      pb[prow * 32 + 16 + (l & 15)] = (_Float16)p1;
    }
    asm volatile("s_wait_dscnt 0" ::: "memory");       // stores visible
    const v16h pa = ldfrag(pb, 32, 0, 0);              // re-read in A layout
    v16h bv0 = ldfrag(vt, NTOK, h * 32, kb);           // V^T: contiguous keys
    v16h bv1 = ldfrag(vt, NTOK, h * 32 + 16, kb);
    oc0 = wmma_f16(pa, bv0, oc0);
    oc1 = wmma_f16(pa, bv1, oc1);
    asm volatile("s_wait_dscnt 0" ::: "memory");       // loads done before rewrite
  }
  const int colb = h * 32 + (l & 15);
  const int rbase = m0 + ((l >> 4) << 3);
#pragma unroll
  for (int j = 0; j < 8; ++j) {
    const float inv = 1.f / lrun[j];
    o[(size_t)(rbase + j) * 64 + colb] = (_Float16)(oc0[j] * inv);
    o[(size_t)(rbase + j) * 64 + colb + 16] = (_Float16)(oc1[j] * inv);
  }
}

// ---------------- driver ----------------

extern "C" void kernel_launch(void* const* d_in, const int* in_sizes, int n_in,
                              void* d_out, int out_size, void* d_ws,
                              size_t ws_size, hipStream_t stream) {
  (void)in_sizes; (void)n_in; (void)out_size; (void)ws_size;
  const int* tok = (const int*)d_in[0];
  const float* embed_w = (const float*)d_in[1];
  const float* pos = (const float*)d_in[2];
  const float* sw = (const float*)d_in[3];
  const float* sb = (const float*)d_in[4];
  const float* ln1g = (const float*)d_in[5];
  const float* ln1b = (const float*)d_in[6];
  const float* wqkv_f = (const float*)d_in[7];
  const float* bqkv = (const float*)d_in[8];
  const float* wo_f = (const float*)d_in[9];
  const float* bo = (const float*)d_in[10];
  const float* ln2g = (const float*)d_in[11];
  const float* ln2b = (const float*)d_in[12];
  const float* w1_f = (const float*)d_in[13];
  const float* b1 = (const float*)d_in[14];
  const float* w2_f = (const float*)d_in[15];
  const float* b2 = (const float*)d_in[16];
  const float* lnfg = (const float*)d_in[17];
  const float* lnfb = (const float*)d_in[18];
  const float* wh_f = (const float*)d_in[19];
  const float* bh = (const float*)d_in[20];

  // workspace layout (all chunks 16B aligned)
  float* x = (float*)d_ws;                 // [4096,64] fp32 master
  float* t = x + NTOK * DMODEL;            // [4096,64] post-attn residual
  float* scv = t + NTOK * DMODEL;          // [4096] sigmoid scores
  _Float16* xl = (_Float16*)(scv + NTOK);  // [4096,64] LN output f16
  _Float16* qkv = xl + NTOK * DMODEL;      // [4096,192]
  _Float16* vt = qkv + NTOK * 192;         // [64,4096] V transposed
  _Float16* oat = vt + DMODEL * NTOK;      // [4096,64] attn output
  _Float16* ffh = oat + NTOK * DMODEL;     // [4096,256]
  _Float16* wqkv = ffh + NTOK * 256;       // f16 weights:
  _Float16* wo = wqkv + NLAYER * 192 * 64;
  _Float16* w1 = wo + NLAYER * 64 * 64;
  _Float16* w2 = w1 + NLAYER * 256 * 64;
  _Float16* wh = w2 + NLAYER * 64 * 256;   // [5008,64] zero-padded

  auto cvt = [&](const float* s, _Float16* dp, int n, int nsrc) {
    k_cvt<<<(n + 255) / 256, 256, 0, stream>>>(s, dp, n, nsrc);
  };
  cvt(wqkv_f, wqkv, NLAYER * 192 * 64, NLAYER * 192 * 64);
  cvt(wo_f, wo, NLAYER * 64 * 64, NLAYER * 64 * 64);
  cvt(w1_f, w1, NLAYER * 256 * 64, NLAYER * 256 * 64);
  cvt(w2_f, w2, NLAYER * 64 * 256, NLAYER * 64 * 256);
  cvt(wh_f, wh, VPAD * 64, NVOCAB * 64);

  k_embed<<<NTOK, 64, 0, stream>>>(tok, embed_w, pos, sw, sb, x, scv);

  for (int li = 0; li < NLAYER; ++li) {
    const float th = (float)li / (float)NLAYER;
    k_ln<<<NTOK, 64, 0, stream>>>(x, ln1g + 64 * li, ln1b + 64 * li, xl);
    k_gemm<0><<<dim3(64, 12), 128, 0, stream>>>(
        xl, wqkv + li * 192 * 64, bqkv + 192 * li, nullptr, qkv, nullptr,
        nullptr, 0.f, 64, 192);
    k_transv<<<NTOK, 64, 0, stream>>>(qkv, vt);
    k_attn<<<128, 128, 0, stream>>>(qkv, vt, scv, th, oat);
    k_gemm<1><<<dim3(64, 4), 128, 0, stream>>>(
        oat, wo + li * 64 * 64, bo + 64 * li, t, nullptr, x, nullptr, 0.f, 64,
        64);
    k_ln<<<NTOK, 64, 0, stream>>>(t, ln2g + 64 * li, ln2b + 64 * li, xl);
    k_gemm<2><<<dim3(64, 16), 128, 0, stream>>>(
        xl, w1 + li * 256 * 64, b1 + 256 * li, nullptr, ffh, nullptr, nullptr,
        0.f, 64, 256);
    k_gemm<3><<<dim3(64, 4), 128, 0, stream>>>(
        ffh, w2 + li * 64 * 256, b2 + 64 * li, x, nullptr, t, scv, th, 256,
        64);
  }
  k_ln<<<NTOK, 64, 0, stream>>>(x, lnfg, lnfb, xl);
  k_gemm<4><<<dim3(64, 313), 128, 0, stream>>>(
      xl, wh, bh, (float*)d_out, nullptr, nullptr, nullptr, 0.f, 64, NVOCAB);
}
